// Simplest_32873679684155
// MI455X (gfx1250) — compile-verified, matrix operand order checked by reading, not run on hardware
//
#include <hip/hip_runtime.h>

#define N_NODES 100000
#define N_EDGES 1600000

typedef __attribute__((ext_vector_type(16))) __bf16        v16bf;
typedef __attribute__((ext_vector_type(8)))  float         v8f;
typedef __attribute__((ext_vector_type(8)))  unsigned int  v8u;
typedef __attribute__((ext_vector_type(4)))  unsigned int  v4u;
typedef __attribute__((ext_vector_type(4)))  unsigned int  u32x4;
typedef __attribute__((ext_vector_type(8)))  int           i32x8;
typedef __attribute__((ext_vector_type(4)))  int           i32x4;

// leaky_relu(x, 0.01) == max(x, 0.01*x)  (exact, since 0 < slope < 1)
__device__ __forceinline__ float lrelu(float x) { return fmaxf(x, 0.01f * x); }

// ---- bf16 fragment loads for v_wmma_f32_16x16x32_bf16 ----
// ISA 7.12.2: lane half selects K octet (+8*half); VGPRs 0-3 hold K pairs
// {0..7}, VGPRs 4-7 hold K pairs {16..23}. Two contiguous 16B runs ->
// 2 x b128 loads per fragment.
__device__ __forceinline__ v16bf ldsFragBF(const __bf16* p, int row, int stride,
                                           int kbase, int half) {
    const v4u* r = (const v4u*)(p + row * stride + kbase + 8 * half);
    v4u lo = r[0];   // elements +0..7
    v4u hi = r[2];   // elements +16..23
    v8u raw;
    raw[0] = lo[0]; raw[1] = lo[1]; raw[2] = lo[2]; raw[3] = lo[3];
    raw[4] = hi[0]; raw[5] = hi[1]; raw[6] = hi[2]; raw[7] = hi[3];
    return __builtin_bit_cast(v16bf, raw);
}

// Same layout, from pre-converted zero-padded bf16 weights in global (L2).
__device__ __forceinline__ v16bf gFragBF(const __bf16* W, int n, int stride,
                                         int kbase, int half) {
    const v4u* r = (const v4u*)(W + (long)n * stride + kbase + 8 * half);
    v4u lo = r[0];
    v4u hi = r[2];
    v8u raw;
    raw[0] = lo[0]; raw[1] = lo[1]; raw[2] = lo[2]; raw[3] = lo[3];
    raw[4] = hi[0]; raw[5] = hi[1]; raw[6] = hi[2]; raw[7] = hi[3];
    return __builtin_bit_cast(v16bf, raw);
}

__device__ __forceinline__ v8f wmma_bf16(v16bf a, v16bf b, v8f c) {
    return __builtin_amdgcn_wmma_f32_16x16x32_bf16(false, a, false, b,
                                                   (short)0, c, false, false);
}

// ---- Tensor Data Mover: 1-D copy global -> LDS (data_size = 8B units) ----
// D# group0: count=1 | lds_addr | global_addr[56:0] | type=2
// D# group1: data_size=3(8B); tensor_dim0 = tile_dim0 = units8B (1-D tile)
__device__ __forceinline__ void tdm_copy_1d(void* lds_ptr, const void* gptr,
                                            unsigned int units8B) {
    unsigned long long ga = (unsigned long long)gptr;
    unsigned int lds_off = (unsigned int)(unsigned long long)lds_ptr; // low32 = LDS offset
    u32x4 g0;
    g0[0] = 1u;                                            // count=1, user mode
    g0[1] = lds_off;                                       // lds_addr
    g0[2] = (unsigned int)(ga & 0xFFFFFFFFu);              // global_addr[31:0]
    g0[3] = (unsigned int)((ga >> 32) & 0x1FFFFFFu) | (2u << 30); // [56:32] | type=2
    i32x8 g1 = {};
    g1[0] = (3 << 16);                                     // data_size = 8B
    g1[1] = (int)((units8B & 0xFFFFu) << 16);              // tensor_dim0[15:0]
    g1[2] = (int)((units8B >> 16) & 0xFFFFu);              // tensor_dim0[31:16]
    g1[3] = (int)((units8B & 0xFFFFu) << 16);              // tile_dim0
    i32x4 z4 = {};
#if __clang_major__ >= 23
    i32x8 z8 = {};
    __builtin_amdgcn_tensor_load_to_lds(g0, g1, z4, z4, z8, 0);
#else
    __builtin_amdgcn_tensor_load_to_lds(g0, g1, z4, z4, 0);
#endif
}

__global__ void zero_kernel(float* __restrict__ p, long n) {
    long i = (long)blockIdx.x * blockDim.x + threadIdx.x;
    long stride = (long)gridDim.x * blockDim.x;
    for (; i < n; i += stride) p[i] = 0.f;
}

// Convert f32 weight [Nsrc][Ksrc] -> zero-padded bf16 [Npad][Kpad] in ws.
__global__ void cvt_pad_kernel(const float* __restrict__ src, __bf16* __restrict__ dst,
                               int Nsrc, int Ksrc, int Npad, int Kpad) {
    int total = Npad * Kpad;
    for (int i = blockIdx.x * blockDim.x + threadIdx.x; i < total;
         i += gridDim.x * blockDim.x) {
        int n = i / Kpad, k = i - n * Kpad;
        float v = (n < Nsrc && k < Ksrc) ? src[n * Ksrc + k] : 0.f;
        dst[i] = (__bf16)v;
    }
}

// ---------------- Edge message MLP fused with scatter-add ----------------
// 256 threads (8 waves), 128 edges per block. N_EDGES % 128 == 0.
__global__ __launch_bounds__(256) void edge_msg_kernel(
    const float* __restrict__ edge_attr, const long long* __restrict__ edge_index,
    const float* __restrict__ Wm1, const float* __restrict__ bm1,
    const __bf16* __restrict__ wm2b, const float* __restrict__ bm2,
    const __bf16* __restrict__ wm3b, const float* __restrict__ bm3,
    float* __restrict__ aggr) {
    extern __shared__ unsigned char smem[];
    __bf16* sWm2 = (__bf16*)smem;                // 128*128 bf16 (32KB)
    __bf16* sWm3 = sWm2 + 16384;                 // 32KB
    __bf16* sH1  = sWm3 + 16384;                 // 32KB
    __bf16* sH2  = sH1 + 16384;                  // 32KB
    float* sEA  = (float*)(sH2 + 16384);         // 128*5
    float* sWm1 = sEA + 640;                     // 128*5
    float* sB1  = sWm1 + 640;
    float* sB2  = sB1 + 128;
    float* sB3  = sB2 + 128;
    int*   sDst = (int*)(sB3 + 128);             // 128

    const int tid  = threadIdx.x;
    const int wave = tid >> 5, lane = tid & 31;
    const int half = lane >> 4, lm = lane & 15;
    const long eBase = (long)blockIdx.x * 128;

    // TDM: DMA the pre-converted bf16 weight tiles straight into LDS.
    if (wave == 0) {
        tdm_copy_1d(sWm2, wm2b, 16384 / 4);   // 32KB = 4096 x 8B
        tdm_copy_1d(sWm3, wm3b, 16384 / 4);
        __builtin_amdgcn_s_wait_tensorcnt(0);
    }
    for (int i = tid; i < 640; i += 256) { sWm1[i] = Wm1[i]; sEA[i] = edge_attr[eBase * 5 + i]; }
    if (tid < 128) {
        sB1[tid] = bm1[tid]; sB2[tid] = bm2[tid]; sB3[tid] = bm3[tid];
        sDst[tid] = (int)edge_index[(long)N_EDGES + eBase + tid];   // dst row
    }
    __syncthreads();

    // h1 = leaky(edge_attr @ Wm1^T + bm1)  (K=5, plain VALU FMA)
    {
        int e = tid >> 1, j0 = (tid & 1) * 64;
        for (int j = 0; j < 64; ++j) {
            int jj = j0 + j;
            float a = sB1[jj];
#pragma unroll
            for (int k = 0; k < 5; ++k) a += sEA[e * 5 + k] * sWm1[jj * 5 + k];
            sH1[e * 128 + jj] = (__bf16)lrelu(a);
        }
    }
    __syncthreads();

    const int mrow = wave * 16 + lm;

    // h2 = leaky(h1 @ Wm2^T + bm2): per wave 8 ntiles x 4 ksteps
#pragma unroll 1
    for (int nt = 0; nt < 8; ++nt) {
        v8f acc = {};
#pragma unroll
        for (int kt = 0; kt < 4; ++kt) {
            v16bf a = ldsFragBF(sH1,  mrow,         128, kt * 32, half);
            v16bf b = ldsFragBF(sWm2, nt * 16 + lm, 128, kt * 32, half);
            acc = wmma_bf16(a, b, acc);
        }
        float bias = sB2[nt * 16 + lm];
#pragma unroll
        for (int v = 0; v < 8; ++v) {
            int r = wave * 16 + v + 8 * half;   // C-fragment row
            sH2[r * 128 + nt * 16 + lm] = (__bf16)lrelu(acc[v] + bias);
        }
    }
    __syncthreads();

    // msg = h2 @ Wm3^T + bm3 ; scatter-add to aggr[dst]
#pragma unroll 1
    for (int nt = 0; nt < 8; ++nt) {
        v8f acc = {};
#pragma unroll
        for (int kt = 0; kt < 4; ++kt) {
            v16bf a = ldsFragBF(sH2,  mrow,         128, kt * 32, half);
            v16bf b = ldsFragBF(sWm3, nt * 16 + lm, 128, kt * 32, half);
            acc = wmma_bf16(a, b, acc);
        }
        float bias = sB3[nt * 16 + lm];
        int col = nt * 16 + lm;
#pragma unroll
        for (int v = 0; v < 8; ++v) {
            int el = wave * 16 + v + 8 * half;
            long d = (long)sDst[el];
            __hip_atomic_fetch_add(&aggr[d * 128 + col], acc[v] + bias,
                                   __ATOMIC_RELAXED, __HIP_MEMORY_SCOPE_AGENT);
        }
    }
}

// ---------------- Node update: LayerNorm + 256->214->172->128 MLP ----------------
// 128 threads (4 waves), 64 nodes per block. Weights pre-padded bf16 in ws:
// wu1b [224][256], wu2b [192][224], wu3b [128][192] (pads are exact zeros).
__global__ __launch_bounds__(128) void node_update_kernel(
    const float* __restrict__ x, const float* __restrict__ aggr,
    const float* __restrict__ ln_g, const float* __restrict__ ln_b,
    const __bf16* __restrict__ wu1b, const float* __restrict__ bu1,
    const __bf16* __restrict__ wu2b, const float* __restrict__ bu2,
    const __bf16* __restrict__ wu3b, const float* __restrict__ bu3,
    float* __restrict__ out) {
    extern __shared__ unsigned char smem[];
    __bf16* sCat = (__bf16*)smem;        // 64*256
    __bf16* sU1  = sCat + 64 * 256;      // 64*224
    __bf16* sU2  = sU1 + 64 * 224;       // 64*192
    float* sSum = (float*)(sU2 + 64 * 192);
    float* sSq  = sSum + 128;
    float* sMu  = sSq + 128;
    float* sRs  = sMu + 64;

    const int tid  = threadIdx.x;
    const int wave = tid >> 5, lane = tid & 31;
    const int half = lane >> 4, lm = lane & 15;
    const long base = (long)blockIdx.x * 64;

    // LayerNorm over concat(x, aggr): 2 threads / row
    {
        int r = tid >> 1, hc = tid & 1;
        long node = base + r; if (node >= N_NODES) node = N_NODES - 1;
        const float* src = hc ? (aggr + node * 128) : (x + node * 128);
        float s = 0.f, q = 0.f;
        for (int i = 0; i < 128; ++i) { float v = src[i]; s += v; q += v * v; }
        sSum[tid] = s; sSq[tid] = q;
    }
    __syncthreads();
    if ((tid & 1) == 0) {
        int r = tid >> 1;
        float s  = sSum[tid] + sSum[tid + 1];
        float q  = sSq[tid]  + sSq[tid + 1];
        float mu = s * (1.f / 256.f);
        float var = q * (1.f / 256.f) - mu * mu;
        sMu[r] = mu; sRs[r] = rsqrtf(var + 1e-5f);
    }
    __syncthreads();
    {
        int r = tid >> 1, hc = tid & 1;
        long node = base + r; if (node >= N_NODES) node = N_NODES - 1;
        const float* src = hc ? (aggr + node * 128) : (x + node * 128);
        float mu = sMu[r], rs = sRs[r];
        int c0 = hc * 128;
        for (int i = 0; i < 128; ++i) {
            int c = c0 + i;
            float nv = (src[i] - mu) * rs * ln_g[c] + ln_b[c];
            sCat[r * 256 + c] = (__bf16)nv;
        }
    }
    __syncthreads();

    const int mrow = wave * 16 + lm;

    // u1 = leaky(cat @ Wu1^T + bu1): N pad 224, K=256
#pragma unroll 1
    for (int nt = 0; nt < 14; ++nt) {
        v8f acc = {};
#pragma unroll
        for (int kt = 0; kt < 8; ++kt) {
            v16bf a = ldsFragBF(sCat, mrow, 256, kt * 32, half);
            v16bf b = gFragBF(wu1b, nt * 16 + lm, 256, kt * 32, half);
            acc = wmma_bf16(a, b, acc);
        }
        int n = nt * 16 + lm;
        float bias = (n < 214) ? bu1[n] : 0.f;
#pragma unroll
        for (int v = 0; v < 8; ++v) {
            int r = wave * 16 + v + 8 * half;
            sU1[r * 224 + n] = (__bf16)lrelu(acc[v] + bias);   // pad cols -> exact 0
        }
    }
    __syncthreads();

    // u2 = leaky(u1 @ Wu2^T + bu2): N pad 192, K pad 224
#pragma unroll 1
    for (int nt = 0; nt < 12; ++nt) {
        v8f acc = {};
#pragma unroll
        for (int kt = 0; kt < 7; ++kt) {
            v16bf a = ldsFragBF(sU1, mrow, 224, kt * 32, half);
            v16bf b = gFragBF(wu2b, nt * 16 + lm, 224, kt * 32, half);
            acc = wmma_bf16(a, b, acc);
        }
        int n = nt * 16 + lm;
        float bias = (n < 172) ? bu2[n] : 0.f;
#pragma unroll
        for (int v = 0; v < 8; ++v) {
            int r = wave * 16 + v + 8 * half;
            sU2[r * 192 + n] = (__bf16)lrelu(acc[v] + bias);
        }
    }
    __syncthreads();

    // out = u2 @ Wu3^T + bu3: N=128, K pad 192
#pragma unroll 1
    for (int nt = 0; nt < 8; ++nt) {
        v8f acc = {};
#pragma unroll
        for (int kt = 0; kt < 6; ++kt) {
            v16bf a = ldsFragBF(sU2, mrow, 192, kt * 32, half);
            v16bf b = gFragBF(wu3b, nt * 16 + lm, 192, kt * 32, half);
            acc = wmma_bf16(a, b, acc);
        }
        int n = nt * 16 + lm;
        float bias = bu3[n];
#pragma unroll
        for (int v = 0; v < 8; ++v) {
            long node = base + wave * 16 + v + 8 * half;
            if (node < N_NODES) out[node * 128 + n] = acc[v] + bias;
        }
    }
}

extern "C" void kernel_launch(void* const* d_in, const int* in_sizes, int n_in,
                              void* d_out, int out_size, void* d_ws, size_t ws_size,
                              hipStream_t stream) {
    (void)in_sizes; (void)n_in; (void)out_size; (void)ws_size;
    const float*     x    = (const float*)d_in[0];
    const long long* ei   = (const long long*)d_in[1];     // int64 [2, E]
    const float*     ea   = (const float*)d_in[2];
    const float*     Wm1  = (const float*)d_in[3];
    const float*     bm1  = (const float*)d_in[4];
    const float*     Wm2  = (const float*)d_in[5];
    const float*     bm2  = (const float*)d_in[6];
    const float*     Wm3  = (const float*)d_in[7];
    const float*     bm3  = (const float*)d_in[8];
    const float*     ln_g = (const float*)d_in[9];
    const float*     ln_b = (const float*)d_in[10];
    const float*     Wu1  = (const float*)d_in[11];
    const float*     bu1  = (const float*)d_in[12];
    const float*     Wu2  = (const float*)d_in[13];
    const float*     bu2  = (const float*)d_in[14];
    const float*     Wu3  = (const float*)d_in[15];
    const float*     bu3  = (const float*)d_in[16];
    float* out  = (float*)d_out;

    // ---- workspace layout ----
    float*  aggr = (float*)d_ws;                           // [N_NODES,128] f32 (51.2MB)
    unsigned char* wsb = (unsigned char*)d_ws + (size_t)N_NODES * 128 * sizeof(float);
    __bf16* wm2b = (__bf16*)wsb;                 wsb += 128 * 128 * 2;
    __bf16* wm3b = (__bf16*)wsb;                 wsb += 128 * 128 * 2;
    __bf16* wu1b = (__bf16*)wsb;                 wsb += 224 * 256 * 2;
    __bf16* wu2b = (__bf16*)wsb;                 wsb += 192 * 224 * 2;
    __bf16* wu3b = (__bf16*)wsb;

    const long nAggr = (long)N_NODES * 128;
    zero_kernel<<<512, 256, 0, stream>>>(aggr, nAggr);

    // Pre-convert all weights to zero-padded bf16 (once per launch, tiny).
    cvt_pad_kernel<<<32, 256, 0, stream>>>(Wm2, wm2b, 128, 128, 128, 128);
    cvt_pad_kernel<<<32, 256, 0, stream>>>(Wm3, wm3b, 128, 128, 128, 128);
    cvt_pad_kernel<<<64, 256, 0, stream>>>(Wu1, wu1b, 214, 256, 224, 256);
    cvt_pad_kernel<<<64, 256, 0, stream>>>(Wu2, wu2b, 172, 214, 192, 224);
    cvt_pad_kernel<<<32, 256, 0, stream>>>(Wu3, wu3b, 128, 172, 128, 192);

    const size_t edgeSmem = 4 * 16384 * sizeof(__bf16)
                          + (640 + 640 + 3 * 128) * sizeof(float) + 128 * sizeof(int);
    edge_msg_kernel<<<N_EDGES / 128, 256, edgeSmem, stream>>>(
        ea, ei, Wm1, bm1, wm2b, bm2, wm3b, bm3, aggr);

    const size_t nodeSmem = (64 * 256 + 64 * 224 + 64 * 192) * sizeof(__bf16)
                          + (128 + 128 + 64 + 64) * sizeof(float);
    node_update_kernel<<<(N_NODES + 63) / 64, 128, nodeSmem, stream>>>(
        x, aggr, ln_g, ln_b, wu1b, bu1, wu2b, bu2, wu3b, bu3, out);
}
